// DeepQNetwork_62036507623969
// MI455X (gfx1250) — matrix operand order repaired and checked by reading, not compiled, not run on hardware
//
#include <hip/hip_runtime.h>

// ---------------------------------------------------------------------------
// MoE DQN (8 experts, 6-layer MLP 256->64x4->18) for gfx1250 (MI455X, wave32).
// Rows are bucketed by expert, then each bucket is processed with f16 WMMA
// (v_wmma_f32_16x16x32_f16) tiles of 16 rows per wave, f32 accumulate.
// ---------------------------------------------------------------------------

typedef __attribute__((ext_vector_type(16))) _Float16 v16h;
typedef __attribute__((ext_vector_type(8)))  _Float16 v8h;
typedef __attribute__((ext_vector_type(8)))  float    v8f;

#define NE   8      // experts
#define ND   256    // input dim
#define NH   64     // hidden width
#define NA   18     // actions
#define NAP  32     // actions padded to 2 N-tiles
#define ROWS_PER_BLOCK 128
#define HS   72     // hbuf row stride in halves (72*2B = 144B, 16B aligned, dodges bank repeats)

__global__ void zero_counts_kernel(int* counts) {
    if (threadIdx.x < NE) counts[threadIdx.x] = 0;
}

__global__ void bucket_kernel(const int* __restrict__ rm, int* __restrict__ counts,
                              int* __restrict__ rowidx, int Bn) {
    int i = blockIdx.x * blockDim.x + threadIdx.x;
    if (i >= Bn) return;
    int e = rm[i];
    int pos = atomicAdd(&counts[e], 1);
    rowidx[(size_t)e * Bn + pos] = i;
}

// dst[e][o][i] = (o < OUT) ? f16(src[e][i][o]) : 0   (transpose + pad + convert)
__global__ void cvt_weights_kernel(const float* __restrict__ src, _Float16* __restrict__ dst,
                                   int IN, int OUT, int OUTP) {
    int idx = blockIdx.x * blockDim.x + threadIdx.x;
    int per = OUTP * IN;
    if (idx >= NE * per) return;
    int e = idx / per;
    int r = idx - e * per;
    int o = r / IN;
    int i = r - o * IN;
    dst[idx] = (o < OUT) ? (_Float16)src[((size_t)e * IN + i) * OUT + o] : (_Float16)0.0f;
}

__device__ __forceinline__ v16h mk16(v8h lo, v8h hi) {
    v16h r;
#pragma unroll
    for (int i = 0; i < 8; ++i) { r[i] = lo[i]; r[i + 8] = hi[i]; }
    return r;
}

__global__ __launch_bounds__(256)
void moe_wmma_kernel(const float* __restrict__ state,
                     const int*   __restrict__ counts,
                     const int*   __restrict__ rowidx,
                     const _Float16* __restrict__ wt1,   // [E][64][256]
                     const _Float16* __restrict__ wt2,   // [E][64][64]
                     const _Float16* __restrict__ wt3,
                     const _Float16* __restrict__ wt4,
                     const _Float16* __restrict__ wt5,
                     const _Float16* __restrict__ wt6,   // [E][32][64] (zero padded)
                     const float* __restrict__ b1, const float* __restrict__ b2,
                     const float* __restrict__ b3, const float* __restrict__ b4,
                     const float* __restrict__ b5, const float* __restrict__ b6,
                     float* __restrict__ out, int Bn)
{
    __shared__ _Float16 wbuf[NH * ND];          // 32 KB: current layer's W^T for this expert
    __shared__ _Float16 hbuf[8 * 16 * HS];      // 18 KB: per-wave 16x64 activation tiles

    const int e     = blockIdx.x % NE;
    const int tile  = blockIdx.x / NE;
    const int count = counts[e];
    if (tile * ROWS_PER_BLOCK >= count) return;   // uniform early-out

    const int tid  = threadIdx.x;
    const int wave = tid >> 5;
    const int lane = tid & 31;
    const int kh   = lane >> 4;     // which K half-group this lane feeds
    const int ln   = lane & 15;     // row (A) / column (B,C,D) index within tile

    // ---- stage W1^T (64x256 f16 = 32KB) into LDS cooperatively ----
    {
        const float4* src = reinterpret_cast<const float4*>(wt1 + (size_t)e * NH * ND);
        float4* dst = reinterpret_cast<float4*>(wbuf);
#pragma unroll
        for (int j = 0; j < 8; ++j) dst[tid + j * 256] = src[tid + j * 256];
    }
    // warm L2 with this expert's W2 while we chew on layer 1
    __builtin_prefetch(wt2 + (size_t)e * NH * NH + tid * 16, 0, 3);
    __syncthreads();

    // gathered source row for this lane's A row (row-in-tile == ln for both lane halves)
    const int basePos = tile * ROWS_PER_BLOCK + wave * 16;
    int posA = basePos + ln; if (posA > count - 1) posA = count - 1;
    const int   gA = rowidx[(size_t)e * Bn + posA];
    const float* xr = state + (size_t)gA * ND;

    _Float16* hrow = &hbuf[wave * 16 * HS];
    const v8f vzero = {};
    v8f acc[4];
#pragma unroll
    for (int t = 0; t < 4; ++t) acc[t] = vzero;

    // ================= layer 1: [16x256] x [256x64] =================
#pragma unroll
    for (int c = 0; c < 8; ++c) {
        // A operand: lanes 0-15 hold K {32c+0..15}; lanes 16-31 hold {32c+16..31} (contiguous)
        const int k1 = c * 32 + kh * 16;
        const int k2 = k1 + 8;
        float4 f0 = *reinterpret_cast<const float4*>(xr + k1);
        float4 f1 = *reinterpret_cast<const float4*>(xr + k1 + 4);
        float4 f2 = *reinterpret_cast<const float4*>(xr + k2);
        float4 f3 = *reinterpret_cast<const float4*>(xr + k2 + 4);
        v16h a;
        a[0]=(_Float16)f0.x; a[1]=(_Float16)f0.y; a[2] =(_Float16)f0.z; a[3] =(_Float16)f0.w;
        a[4]=(_Float16)f1.x; a[5]=(_Float16)f1.y; a[6] =(_Float16)f1.z; a[7] =(_Float16)f1.w;
        a[8]=(_Float16)f2.x; a[9]=(_Float16)f2.y; a[10]=(_Float16)f2.z; a[11]=(_Float16)f2.w;
        a[12]=(_Float16)f3.x;a[13]=(_Float16)f3.y;a[14]=(_Float16)f3.z; a[15]=(_Float16)f3.w;
#pragma unroll
        for (int t = 0; t < 4; ++t) {
            // B operand: lane n<16 -> col n, K {32c+0..15}; lane>=16 -> K {32c+16..31} (contiguous)
            const _Float16* wp = &wbuf[(t * 16 + ln) * ND + c * 32 + kh * 16];
            v16h b = *reinterpret_cast<const v16h*>(wp);
            acc[t] = __builtin_amdgcn_wmma_f32_16x16x32_f16(false, a, false, b,
                                                            (short)0, acc[t], false, false);
        }
    }
    // bias + ReLU + lane-transpose via LDS (D: VGPR i -> row i + 8*kh, col = ln)
#pragma unroll
    for (int t = 0; t < 4; ++t) {
        const int col = t * 16 + ln;
        const float bv = b1[e * NH + col];
#pragma unroll
        for (int i = 0; i < 8; ++i) {
            float v = acc[t][i] + bv;
            v = v > 0.0f ? v : 0.0f;
            hrow[(i + kh * 8) * HS + col] = (_Float16)v;
        }
    }

    // ================= layers 2..5: [16x64] x [64x64] =================
    const _Float16* wts[4] = { wt2, wt3, wt4, wt5 };
    const float*    bs[4]  = { b2, b3, b4, b5 };
    for (int l = 0; l < 4; ++l) {
        __syncthreads();   // everyone finished reading previous weights
        {
            const float4* src = reinterpret_cast<const float4*>(wts[l] + (size_t)e * NH * NH);
            float4* dst = reinterpret_cast<float4*>(wbuf);
            dst[tid]       = src[tid];
            dst[tid + 256] = src[tid + 256];
        }
        __builtin_prefetch(((l < 3) ? wts[l + 1] + (size_t)e * NH * NH
                                    : wt6 + (size_t)e * NAP * NH) + tid * 8, 0, 3);
        __syncthreads();

#pragma unroll
        for (int t = 0; t < 4; ++t) acc[t] = vzero;
#pragma unroll
        for (int c = 0; c < 2; ++c) {
            const int k1 = c * 32 + kh * 16;
            const _Float16* ap = &hrow[ln * HS + k1];
            v8h alo = *reinterpret_cast<const v8h*>(ap);
            v8h ahi = *reinterpret_cast<const v8h*>(ap + 8);
            v16h a  = mk16(alo, ahi);
#pragma unroll
            for (int t = 0; t < 4; ++t) {
                const _Float16* wp = &wbuf[(t * 16 + ln) * NH + c * 32 + kh * 16];
                v16h b = *reinterpret_cast<const v16h*>(wp);
                acc[t] = __builtin_amdgcn_wmma_f32_16x16x32_f16(false, a, false, b,
                                                                (short)0, acc[t], false, false);
            }
        }
        const float* bl = bs[l];
#pragma unroll
        for (int t = 0; t < 4; ++t) {
            const int col = t * 16 + ln;
            const float bv = bl[e * NH + col];
#pragma unroll
            for (int i = 0; i < 8; ++i) {
                float v = acc[t][i] + bv;
                v = v > 0.0f ? v : 0.0f;
                hrow[(i + kh * 8) * HS + col] = (_Float16)v;   // in-place: all A reads done above
            }
        }
    }

    // ================= layer 6 (head): [16x64] x [64x32(pad of 18)] =================
    __syncthreads();
    {
        const float4* src = reinterpret_cast<const float4*>(wt6 + (size_t)e * NAP * NH);
        float4* dst = reinterpret_cast<float4*>(wbuf);
        dst[tid] = src[tid];   // 32*64 f16 = 4KB = 256 float4
    }
    __syncthreads();

    v8f acc2[2] = { vzero, vzero };
#pragma unroll
    for (int c = 0; c < 2; ++c) {
        const int k1 = c * 32 + kh * 16;
        const _Float16* ap = &hrow[ln * HS + k1];
        v8h alo = *reinterpret_cast<const v8h*>(ap);
        v8h ahi = *reinterpret_cast<const v8h*>(ap + 8);
        v16h a  = mk16(alo, ahi);
#pragma unroll
        for (int t = 0; t < 2; ++t) {
            const _Float16* wp = &wbuf[(t * 16 + ln) * NH + c * 32 + kh * 16];
            v16h b = *reinterpret_cast<const v16h*>(wp);
            acc2[t] = __builtin_amdgcn_wmma_f32_16x16x32_f16(false, a, false, b,
                                                             (short)0, acc2[t], false, false);
        }
    }

    // scatter actions back to their original rows (all WMMA work is done; masking is safe now)
#pragma unroll
    for (int t = 0; t < 2; ++t) {
        const int col = t * 16 + ln;
        if (col >= NA) continue;
        const float bv = b6[e * NA + col];
#pragma unroll
        for (int i = 0; i < 8; ++i) {
            const int m = i + kh * 8;
            const int pos = basePos + m;
            if (pos < count) {
                const int g = rowidx[(size_t)e * Bn + pos];
                out[(size_t)g * NA + col] = acc2[t][i] + bv;
            }
        }
    }
}

extern "C" void kernel_launch(void* const* d_in, const int* in_sizes, int n_in,
                              void* d_out, int out_size, void* d_ws, size_t ws_size,
                              hipStream_t stream) {
    (void)n_in; (void)out_size; (void)ws_size;
    const float* state = (const float*)d_in[0];
    const int*   rm    = (const int*)d_in[1];
    const float* W1 = (const float*)d_in[2];  const float* b1 = (const float*)d_in[3];
    const float* W2 = (const float*)d_in[4];  const float* b2 = (const float*)d_in[5];
    const float* W3 = (const float*)d_in[6];  const float* b3 = (const float*)d_in[7];
    const float* W4 = (const float*)d_in[8];  const float* b4 = (const float*)d_in[9];
    const float* W5 = (const float*)d_in[10]; const float* b5 = (const float*)d_in[11];
    const float* W6 = (const float*)d_in[12]; const float* b6 = (const float*)d_in[13];
    float* out = (float*)d_out;
    const int Bn = in_sizes[1];   // rm_state element count == batch

    // ---- workspace layout (all 256B aligned) ----
    char* ws = (char*)d_ws;
    int* counts = (int*)ws;                                   // 8 ints (pad 256)
    int* rowidx = (int*)(ws + 256);                           // E*Bn ints
    size_t off = 256 + (size_t)NE * Bn * sizeof(int);
    off = (off + 255) & ~(size_t)255;
    _Float16* wt1 = (_Float16*)(ws + off); off += (size_t)NE * NH  * ND * 2;
    _Float16* wt2 = (_Float16*)(ws + off); off += (size_t)NE * NH  * NH * 2;
    _Float16* wt3 = (_Float16*)(ws + off); off += (size_t)NE * NH  * NH * 2;
    _Float16* wt4 = (_Float16*)(ws + off); off += (size_t)NE * NH  * NH * 2;
    _Float16* wt5 = (_Float16*)(ws + off); off += (size_t)NE * NH  * NH * 2;
    _Float16* wt6 = (_Float16*)(ws + off); off += (size_t)NE * NAP * NH * 2;

    // pass 0/1: bucket rows by expert, convert+transpose weights to f16
    zero_counts_kernel<<<1, 32, 0, stream>>>(counts);
    bucket_kernel<<<(Bn + 255) / 256, 256, 0, stream>>>(rm, counts, rowidx, Bn);

    cvt_weights_kernel<<<(NE * NH  * ND + 255) / 256, 256, 0, stream>>>(W1, wt1, ND, NH, NH);
    cvt_weights_kernel<<<(NE * NH  * NH + 255) / 256, 256, 0, stream>>>(W2, wt2, NH, NH, NH);
    cvt_weights_kernel<<<(NE * NH  * NH + 255) / 256, 256, 0, stream>>>(W3, wt3, NH, NH, NH);
    cvt_weights_kernel<<<(NE * NH  * NH + 255) / 256, 256, 0, stream>>>(W4, wt4, NH, NH, NH);
    cvt_weights_kernel<<<(NE * NH  * NH + 255) / 256, 256, 0, stream>>>(W5, wt5, NH, NH, NH);
    cvt_weights_kernel<<<(NE * NAP * NH + 255) / 256, 256, 0, stream>>>(W6, wt6, NH, NA, NAP);

    // pass 2: WMMA MoE MLP. Round-robin experts across blockIdx so empty tiles spread out.
    const int tilesPerE = (Bn + ROWS_PER_BLOCK - 1) / ROWS_PER_BLOCK;
    moe_wmma_kernel<<<NE * tilesPerE, 256, 0, stream>>>(
        state, counts, rowidx, wt1, wt2, wt3, wt4, wt5, wt6,
        b1, b2, b3, b4, b5, b6, out, Bn);
}
